// Decoder_45303315038439
// MI455X (gfx1250) — matrix operand order repaired and checked by reading, not compiled, not run on hardware
//
#include <hip/hip_runtime.h>
#include <hip/hip_bf16.h>
#include <math.h>

#define EMBED   128
#define HIDDEN  256
#define GATES   1024   // 4*HIDDEN
#define VOCAB   410
#define VOCABP  448    // padded to 7 tiles of 64
#define SEQ     100
#define TSTEPS  99
#define BATCH   1024

#define SROW    40     // padded LDS row for 32-K chunks (80B: 16B-aligned, bank-conflict-free)
#define LROW    264    // padded LDS row for 256-K strips (528B)

typedef __bf16 bf16_t;
typedef __attribute__((ext_vector_type(16))) __bf16 bf16x16;
typedef __attribute__((ext_vector_type(8)))  __bf16 bf16x8;
typedef __attribute__((ext_vector_type(8)))  float  f32x8;

// ---------------------------------------------------------------------------
// WMMA helpers (CDNA5 wave32, 16x16x32 bf16 -> f32)
// ---------------------------------------------------------------------------
static __device__ __forceinline__ f32x8 wmma_bf16(bf16x16 a, bf16x16 b, f32x8 c) {
  return __builtin_amdgcn_wmma_f32_16x16x32_bf16(false, a, false, b, (short)0, c,
                                                 false, false);
}

// A tile 16x32 (row-major, rows base..base+15, cols k0..k0+31).
// lanes 0-15: M=lane, elems 0-15=K0-15; lanes 16-31: K16-31.
static __device__ __forceinline__ bf16x16
load_a_tile(const bf16_t* base, int lda, int lane, int k0) {
  const bf16_t* p = base + (size_t)(lane & 15) * lda + k0 + ((lane >> 4) << 4);
  bf16x8 lo = *(const bf16x8*)p;
  bf16x8 hi = *(const bf16x8*)(p + 8);
  return __builtin_shufflevector(lo, hi, 0, 1, 2, 3, 4, 5, 6, 7,
                                 8, 9, 10, 11, 12, 13, 14, 15);
}

// A tile with per-lane gathered row pointer (embedding / hs gather).
static __device__ __forceinline__ bf16x16
load_a_gather(const bf16_t* rowp, int lane, int k0) {
  const bf16_t* p = rowp + k0 + ((lane >> 4) << 4);
  bf16x8 lo = *(const bf16x8*)p;
  bf16x8 hi = *(const bf16x8*)(p + 8);
  return __builtin_shufflevector(lo, hi, 0, 1, 2, 3, 4, 5, 6, 7,
                                 8, 9, 10, 11, 12, 13, 14, 15);
}

// B tile 32x16 direct from global (weights (out,in) row-major; column n = row n).
static __device__ __forceinline__ bf16x16
load_b_tile(const bf16_t* W, int ldb, int n0, int lane, int k0) {
  const bf16_t* p = W + (size_t)(n0 + (lane & 15)) * ldb + k0 + ((lane >> 4) << 4);
  return *(const bf16x16*)p;
}

// B tile 32x16 from an LDS chunk laid out [strip_col][SROW k-elems].
static __device__ __forceinline__ bf16x16
load_b_lds(const bf16_t* sbuf, int c, int lane) {
  return *(const bf16x16*)(sbuf + c * SROW + ((lane >> 4) << 4));
}

// CDNA5 async global->LDS copy of 16B (tracked by ASYNCcnt).
static __device__ __forceinline__ void
async_b128(uint32_t lds_addr, const bf16_t* gaddr) {
  asm volatile("global_load_async_to_lds_b128 %0, %1, off"
               :: "v"(lds_addr), "v"(gaddr) : "memory");
}

// Stage one 256-col x 32-K weight chunk into LDS (one col per thread, 4x16B).
static __device__ __forceinline__ void
stage_strip_chunk(const bf16_t* __restrict__ W, int ldb, int j0, int k0,
                  bf16_t* sbuf, int tid) {
  const int gate = tid >> 6, jl = tid & 63;
  const bf16_t* src = W + (size_t)(gate * HIDDEN + j0 + jl) * ldb + k0;
  const uint32_t dst = (uint32_t)(uintptr_t)(sbuf + tid * SROW);
#pragma unroll
  for (int i = 0; i < 4; ++i)
    async_b128(dst + i * 16, src + i * 8);
}

static __device__ __forceinline__ float sigf(float x) {
  return 1.0f / (1.0f + __expf(-x));
}

// ---------------------------------------------------------------------------
// Small setup kernels
// ---------------------------------------------------------------------------
__global__ void k_f32_to_bf16(const float* __restrict__ s, bf16_t* __restrict__ d, int n) {
  for (int i = blockIdx.x * blockDim.x + threadIdx.x; i < n; i += gridDim.x * blockDim.x)
    d[i] = (bf16_t)s[i];
}
__global__ void k_zero_bf16(bf16_t* d, int n) {
  for (int i = blockIdx.x * blockDim.x + threadIdx.x; i < n; i += gridDim.x * blockDim.x)
    d[i] = (bf16_t)0.0f;
}
__global__ void k_zero_f32(float* d, int n) {
  for (int i = blockIdx.x * blockDim.x + threadIdx.x; i < n; i += gridDim.x * blockDim.x)
    d[i] = 0.0f;
}
__global__ void k_bias_sum(const float* a, const float* b, float* o, int n) {
  for (int i = blockIdx.x * blockDim.x + threadIdx.x; i < n; i += gridDim.x * blockDim.x)
    o[i] = a[i] + b[i];
}

// ---------------------------------------------------------------------------
// enc = encoder_out @ W_enc^T + b_enc (run once) -> bf16. grid (8,4)
// ---------------------------------------------------------------------------
__global__ __launch_bounds__(256) void k_enc(const bf16_t* __restrict__ A,
                                             const bf16_t* __restrict__ W,
                                             const float* __restrict__ bias,
                                             bf16_t* __restrict__ out) {
  const int lane = threadIdx.x & 31, wave = threadIdx.x >> 5;
  const int m0 = blockIdx.x * 128 + wave * 16;
  const int n0 = blockIdx.y * 64;
  f32x8 acc[4] = {};
  for (int k0 = 0; k0 < HIDDEN; k0 += 32) {
    bf16x16 a = load_a_tile(A + (size_t)m0 * HIDDEN, HIDDEN, lane, k0);
#pragma unroll
    for (int nt = 0; nt < 4; ++nt)
      acc[nt] = wmma_bf16(a, load_b_tile(W, HIDDEN, n0 + nt * 16, lane, k0), acc[nt]);
  }
  const int mb = (lane >> 4) << 3, nc = lane & 15;
#pragma unroll
  for (int nt = 0; nt < 4; ++nt)
#pragma unroll
    for (int r = 0; r < 8; ++r) {
      int n = n0 + nt * 16 + nc;
      out[(size_t)(m0 + mb + r) * HIDDEN + n] = (bf16_t)(acc[nt][r] + bias[n]);
    }
}

// ---------------------------------------------------------------------------
// genc = enc @ W_ih[:,128:]^T + (b_ih+b_hh) (run once) -> f32. grid (8,16)
// ---------------------------------------------------------------------------
__global__ __launch_bounds__(256) void k_genc(const bf16_t* __restrict__ A,
                                              const bf16_t* __restrict__ Wih,
                                              const float* __restrict__ bias,
                                              float* __restrict__ out) {
  const int lane = threadIdx.x & 31, wave = threadIdx.x >> 5;
  const int m0 = blockIdx.x * 128 + wave * 16;
  const int n0 = blockIdx.y * 64;
  const bf16_t* B = Wih + EMBED;  // hidden half of W_ih rows (ld = 384)
  f32x8 acc[4] = {};
  for (int k0 = 0; k0 < HIDDEN; k0 += 32) {
    bf16x16 a = load_a_tile(A + (size_t)m0 * HIDDEN, HIDDEN, lane, k0);
#pragma unroll
    for (int nt = 0; nt < 4; ++nt)
      acc[nt] = wmma_bf16(a, load_b_tile(B, EMBED + HIDDEN, n0 + nt * 16, lane, k0), acc[nt]);
  }
  const int mb = (lane >> 4) << 3, nc = lane & 15;
#pragma unroll
  for (int nt = 0; nt < 4; ++nt)
#pragma unroll
    for (int r = 0; r < 8; ++r) {
      int n = n0 + nt * 16 + nc;
      out[(size_t)(m0 + mb + r) * GATES + n] = acc[nt][r] + bias[n];
    }
}

// ---------------------------------------------------------------------------
// One LSTM step (99 launches):
//   gates = h_{t-1} @ W_hh^T + emb[tgt[:,t]] @ W_ih[:,:128]^T + genc
//   c = sig(f)*c + sig(i)*tanh(g);  h_t = sig(o)*tanh(c)
// Block 256 thr = 8 waves on batch: tile 128b x 64j x 4 gates.
// Weight strips stream through double-buffered LDS via async-to-LDS copies.
// ---------------------------------------------------------------------------
__global__ __launch_bounds__(256) void k_step(bf16_t* __restrict__ h_all,
                                              const bf16_t* __restrict__ emb,
                                              const bf16_t* __restrict__ Whh,
                                              const bf16_t* __restrict__ Wih,
                                              const int* __restrict__ tgt,
                                              const float* __restrict__ genc,
                                              float* __restrict__ cbuf,
                                              int t) {
  __shared__ bf16_t sB[2][256 * SROW];  // 2 x 20KB chunks
  const int tid = threadIdx.x;
  const int lane = tid & 31, wave = tid >> 5;
  const int b0 = blockIdx.x * 128 + wave * 16;
  const int j0 = blockIdx.y * 64;

  f32x8 acc[4][4] = {};  // [gate][ntile]
  const bf16_t* hprev = h_all + ((size_t)t * BATCH + b0) * HIDDEN;
  const int brow = b0 + (lane & 15);
  const bf16_t* erow = emb + (size_t)tgt[brow * SEQ + t] * EMBED;

  // ---- phase 1: recurrent K=256 over h_{t-1}, 8 chunks, double buffered
  stage_strip_chunk(Whh, HIDDEN, j0, 0, sB[0], tid);
  stage_strip_chunk(Whh, HIDDEN, j0, 32, sB[1], tid);
#pragma unroll
  for (int kc = 0; kc < 8; ++kc) {
    if (kc < 7) asm volatile("s_wait_asynccnt 0x4" ::: "memory");
    else        asm volatile("s_wait_asynccnt 0x0" ::: "memory");
    __syncthreads();
    bf16x16 a = load_a_tile(hprev, HIDDEN, lane, kc * 32);
    const bf16_t* sb = sB[kc & 1];
#pragma unroll
    for (int g = 0; g < 4; ++g)
#pragma unroll
      for (int nt = 0; nt < 4; ++nt)
        acc[g][nt] = wmma_bf16(a, load_b_lds(sb, g * 64 + nt * 16 + (lane & 15), lane),
                               acc[g][nt]);
    __syncthreads();
    if (kc + 2 < 8)
      stage_strip_chunk(Whh, HIDDEN, j0, (kc + 2) * 32, sB[kc & 1], tid);
  }

  // ---- phase 2: embedding K=128, 4 chunks, A rows gathered via targets[:,t]
  stage_strip_chunk(Wih, EMBED + HIDDEN, j0, 0, sB[0], tid);
  stage_strip_chunk(Wih, EMBED + HIDDEN, j0, 32, sB[1], tid);
#pragma unroll
  for (int kc = 0; kc < 4; ++kc) {
    if (kc < 3) asm volatile("s_wait_asynccnt 0x4" ::: "memory");
    else        asm volatile("s_wait_asynccnt 0x0" ::: "memory");
    __syncthreads();
    bf16x16 a = load_a_gather(erow, lane, kc * 32);
    const bf16_t* sb = sB[kc & 1];
#pragma unroll
    for (int g = 0; g < 4; ++g)
#pragma unroll
      for (int nt = 0; nt < 4; ++nt)
        acc[g][nt] = wmma_bf16(a, load_b_lds(sb, g * 64 + nt * 16 + (lane & 15), lane),
                               acc[g][nt]);
    __syncthreads();
    if (kc + 2 < 4)
      stage_strip_chunk(Wih, EMBED + HIDDEN, j0, (kc + 2) * 32, sB[kc & 1], tid);
  }

  // ---- LSTM cell (all 4 gates for (b,j) live in this lane)
  bf16_t* hnext = h_all + (size_t)(t + 1) * BATCH * HIDDEN;
  const int mb = (lane >> 4) << 3, nc = lane & 15;
#pragma unroll
  for (int nt = 0; nt < 4; ++nt)
#pragma unroll
    for (int r = 0; r < 8; ++r) {
      const int b = b0 + mb + r;
      const int j = j0 + nt * 16 + nc;
      const float* gp = genc + (size_t)b * GATES + j;
      float iv = sigf(acc[0][nt][r] + gp[0]);
      float fv = sigf(acc[1][nt][r] + gp[HIDDEN]);
      float gv = tanhf(acc[2][nt][r] + gp[2 * HIDDEN]);
      float ov = sigf(acc[3][nt][r] + gp[3 * HIDDEN]);
      float c = fv * cbuf[(size_t)b * HIDDEN + j] + iv * gv;
      cbuf[(size_t)b * HIDDEN + j] = c;
      hnext[(size_t)b * HIDDEN + j] = (bf16_t)(ov * tanhf(c));
    }
}

// ---------------------------------------------------------------------------
// logits = hs @ W_fc^T + b_fc  ((1024*99)x256 @ 256x410). grid (792,7).
// Whole 64-col x 256-K weight strip staged once in LDS via async copies.
// ---------------------------------------------------------------------------
__global__ __launch_bounds__(256) void k_logits(const bf16_t* __restrict__ h_all,
                                                const bf16_t* __restrict__ Wfc,
                                                const float* __restrict__ bias,
                                                float* __restrict__ out) {
  __shared__ bf16_t sW[64 * LROW];  // ~33KB
  const int tid = threadIdx.x;
  const int lane = tid & 31, wave = tid >> 5;
  const int m0 = blockIdx.x * 128 + wave * 16;
  const int n0 = blockIdx.y * 64;

  {  // cooperative strip stage: 64 cols x 512B, 128B per thread
    const int c = tid >> 2, q = tid & 3;
    const bf16_t* src = Wfc + (size_t)(n0 + c) * HIDDEN + q * 64;
    const uint32_t dst = (uint32_t)(uintptr_t)(sW + c * LROW + q * 64);
#pragma unroll
    for (int i = 0; i < 8; ++i)
      async_b128(dst + i * 16, src + i * 8);
  }
  asm volatile("s_wait_asynccnt 0x0" ::: "memory");
  __syncthreads();

  const int m = m0 + (lane & 15);
  const int b = m / TSTEPS, tt = m - b * TSTEPS;
  const bf16_t* arow = h_all + ((size_t)(tt + 1) * BATCH + b) * HIDDEN;

  f32x8 acc[4] = {};
  for (int k0 = 0; k0 < HIDDEN; k0 += 32) {
    bf16x16 a = load_a_gather(arow, lane, k0);
#pragma unroll
    for (int nt = 0; nt < 4; ++nt) {
      bf16x16 bt = *(const bf16x16*)(sW + (nt * 16 + (lane & 15)) * LROW + k0 +
                                     ((lane >> 4) << 4));
      acc[nt] = wmma_bf16(a, bt, acc[nt]);
    }
  }
  const int mb = (lane >> 4) << 3, nc = lane & 15;
#pragma unroll
  for (int nt = 0; nt < 4; ++nt)
#pragma unroll
    for (int r = 0; r < 8; ++r) {
      const int mm = m0 + mb + r;
      const int bb = mm / TSTEPS, t2 = mm - bb * TSTEPS;
      const int n = n0 + nt * 16 + nc;
      if (n < VOCAB)
        out[((size_t)bb * TSTEPS + t2) * VOCAB + n] = acc[nt][r] + bias[n];
    }
}

// ---------------------------------------------------------------------------
extern "C" void kernel_launch(void* const* d_in, const int* in_sizes, int n_in,
                              void* d_out, int out_size, void* d_ws, size_t ws_size,
                              hipStream_t stream) {
  (void)in_sizes; (void)n_in; (void)out_size; (void)ws_size;
  const float* encoder_out = (const float*)d_in[0];
  const int*   targets     = (const int*)d_in[1];
  const float* emb         = (const float*)d_in[2];
  const float* W_enc       = (const float*)d_in[3];
  const float* b_enc       = (const float*)d_in[4];
  const float* W_ih        = (const float*)d_in[5];
  const float* W_hh        = (const float*)d_in[6];
  const float* b_ih        = (const float*)d_in[7];
  const float* b_hh        = (const float*)d_in[8];
  const float* W_fc        = (const float*)d_in[9];
  const float* b_fc        = (const float*)d_in[10];
  float* out = (float*)d_out;

  // ---- workspace carve-up (256-B aligned slabs, ~61 MB total)
  char* ws = (char*)d_ws;
  size_t off = 0;
  auto carve = [&](size_t bytes) {
    char* p = ws + off;
    off = (off + bytes + 255) & ~(size_t)255;
    return p;
  };
  bf16_t* w_enc_bf = (bf16_t*)carve((size_t)HIDDEN * HIDDEN * 2);
  bf16_t* w_ih_bf  = (bf16_t*)carve((size_t)GATES * (EMBED + HIDDEN) * 2);
  bf16_t* w_hh_bf  = (bf16_t*)carve((size_t)GATES * HIDDEN * 2);
  bf16_t* w_fc_bf  = (bf16_t*)carve((size_t)VOCABP * HIDDEN * 2);
  bf16_t* emb_bf   = (bf16_t*)carve((size_t)VOCAB * EMBED * 2);
  bf16_t* encin_bf = (bf16_t*)carve((size_t)BATCH * HIDDEN * 2);
  bf16_t* enc_bf   = (bf16_t*)carve((size_t)BATCH * HIDDEN * 2);
  float*  genc     = (float*)carve((size_t)BATCH * GATES * 4);
  float*  cbuf     = (float*)carve((size_t)BATCH * HIDDEN * 4);
  float*  bias_c   = (float*)carve((size_t)GATES * 4);
  bf16_t* h_all    = (bf16_t*)carve((size_t)SEQ * BATCH * HIDDEN * 2);  // slot 0 = h0

  const int TPB = 256;
  auto blks = [](int n) { return (n + 255) / 256; };

  // ---- setup: bf16 weight/activation images, bias sum, state zeroing
  k_f32_to_bf16<<<blks(HIDDEN * HIDDEN), TPB, 0, stream>>>(W_enc, w_enc_bf, HIDDEN * HIDDEN);
  k_f32_to_bf16<<<blks(GATES * (EMBED + HIDDEN)), TPB, 0, stream>>>(W_ih, w_ih_bf, GATES * (EMBED + HIDDEN));
  k_f32_to_bf16<<<blks(GATES * HIDDEN), TPB, 0, stream>>>(W_hh, w_hh_bf, GATES * HIDDEN);
  k_zero_bf16<<<blks(VOCABP * HIDDEN), TPB, 0, stream>>>(w_fc_bf, VOCABP * HIDDEN);
  k_f32_to_bf16<<<blks(VOCAB * HIDDEN), TPB, 0, stream>>>(W_fc, w_fc_bf, VOCAB * HIDDEN);
  k_f32_to_bf16<<<blks(VOCAB * EMBED), TPB, 0, stream>>>(emb, emb_bf, VOCAB * EMBED);
  k_f32_to_bf16<<<blks(BATCH * HIDDEN), TPB, 0, stream>>>(encoder_out, encin_bf, BATCH * HIDDEN);
  k_bias_sum<<<blks(GATES), TPB, 0, stream>>>(b_ih, b_hh, bias_c, GATES);
  k_zero_f32<<<blks(BATCH * HIDDEN), TPB, 0, stream>>>(cbuf, BATCH * HIDDEN);
  k_zero_bf16<<<blks(BATCH * HIDDEN), TPB, 0, stream>>>(h_all, BATCH * HIDDEN);

  // ---- enc projection, then t-invariant gate contribution
  k_enc<<<dim3(BATCH / 128, HIDDEN / 64), TPB, 0, stream>>>(encin_bf, w_enc_bf, b_enc, enc_bf);
  k_genc<<<dim3(BATCH / 128, GATES / 64), TPB, 0, stream>>>(enc_bf, w_ih_bf, bias_c, genc);

  // ---- sequential LSTM scan: 99 fused GEMM+cell steps
  for (int t = 0; t < TSTEPS; ++t)
    k_step<<<dim3(BATCH / 128, HIDDEN / 64), TPB, 0, stream>>>(
        h_all, emb_bf, w_hh_bf, w_ih_bf, targets, genc, cbuf, t);

  // ---- output projection over all (b,t)
  k_logits<<<dim3((BATCH * TSTEPS) / 128, VOCABP / 64), TPB, 0, stream>>>(
      h_all, w_fc_bf, b_fc, out);
}
